// EGNNLayer_84524956385321
// MI455X (gfx1250) — matrix operand order repaired and checked by reading, not compiled, not run on hardware
//
#include <hip/hip_runtime.h>
#include <hip/hip_bf16.h>

// ---------------------------------------------------------------------------
// EGNN layer for MI455X (gfx1250, wave32, WMMA).
// Edge MLP + gate + segment-sum via f32 atomics, node-update MLP + LayerNorm.
// All GEMMs run on v_wmma_f32_16x16x32_bf16 (bf16 in, f32 accumulate).
// ---------------------------------------------------------------------------

#define N_SRC  50000
#define N_TGT  50000
#define N_EDGE 800000
#define CCH    128
#define K1PAD  416   // 385 msg features padded to multiple of 32

typedef __attribute__((ext_vector_type(16))) __bf16 v16bf;
typedef __attribute__((ext_vector_type(8)))  float  v8f;

union AFrag { v16bf v; __bf16 e[16]; };
union Acc   { v8f   v; float  e[8];  };

__device__ __forceinline__ v8f wmma_bf16(v16bf a, v16bf b, v8f c) {
  // (neg_a, A, neg_b, B, c_mod, C, reuse_a, reuse_b)
  return __builtin_amdgcn_wmma_f32_16x16x32_bf16(false, a, false, b,
                                                 (short)0, c, false, false);
}
__device__ __forceinline__ float siluf(float x) {
  return x * (1.0f / (1.0f + __expf(-x)));
}
// 16-bit A-matrix 16x32 lane/half -> K mapping (matches B staging):
// lanes 0-15 hold K 0..15, lanes 16-31 hold K 16..31
__device__ __forceinline__ int akmap(int lane, int h) {
  return h + ((lane >> 4) << 4);
}
// 16-bit B-matrix 32x16, this code's k order (same as akmap, not the ISA table): lanes 0-15 K=0..15 (2 per VGPR), lanes 16-31 K=16..31
__device__ __forceinline__ int bkmap(int lane, int h) {
  return h + ((lane >> 4) << 4);
}

// ---------------------------------------------------------------------------
// Edge kernel: 64 edges per block, 4 waves, one 16-edge M-tile per wave.
//   GEMM1: [64 x 416] @ [416 x 256] -> silu -> hid
//   GEMM2: [64 x 256] @ [256 x 128] -> silu -> msg  (atomic agg scatter)
//   GEMM3: [64 x 128] @ [128 x 64]  -> silu -> gate hidden
//   gate:  dot with W_c2 -> tanh -> w ; atomic vel scatter of w*rel
// ---------------------------------------------------------------------------
__global__ __launch_bounds__(128) void egnn_edge_kernel(
    const float* __restrict__ h_src, const float* __restrict__ h_tgt,
    const float* __restrict__ pos_src, const float* __restrict__ pos_tgt,
    const float* __restrict__ t_emb,
    const __bf16* __restrict__ wm1, const float* __restrict__ bm1,
    const __bf16* __restrict__ wm2, const float* __restrict__ bm2,
    const __bf16* __restrict__ wc1, const float* __restrict__ bc1,
    const float* __restrict__ wc2, const float* __restrict__ bc2,
    const int* __restrict__ edge_src, const int* __restrict__ edge_tgt,
    float* __restrict__ agg, float* __restrict__ vel)
{
  // region0: B slabs (fragment-swizzled), later the bf16 msg tile (64x128)
  __shared__ __align__(32) __bf16 sB[8192];            // 16 KB
  // region1: hid bf16 (64x256), later gate hidden f32 (64x64)
  __shared__ __align__(32) unsigned char sR1raw[32768]; // 32 KB
  __shared__ int   sEs[64], sEt[64];
  __shared__ float sSq[64], sRel[192];

  const int tid    = threadIdx.x;
  const int lane   = tid & 31;
  const int wave   = tid >> 5;
  const int laneLo = lane & 15;
  const int mBase  = ((lane >> 4) << 3);     // C/D layout M offset (0 or 8)
  const int rowL   = wave * 16 + laneLo;     // A-fragment row (local)
  const int e0     = blockIdx.x * 64;        // N_EDGE % 64 == 0

  // ---- per-edge metadata -------------------------------------------------
  if (tid < 64) {
    int e  = e0 + tid;
    int es = edge_src[e], et = edge_tgt[e];
    float rx = pos_tgt[et * 3 + 0] - pos_src[es * 3 + 0];
    float ry = pos_tgt[et * 3 + 1] - pos_src[es * 3 + 1];
    float rz = pos_tgt[et * 3 + 2] - pos_src[es * 3 + 2];
    sEs[tid] = es; sEt[tid] = et;
    sRel[tid * 3 + 0] = rx; sRel[tid * 3 + 1] = ry; sRel[tid * 3 + 2] = rz;
    sSq[tid] = rx * rx + ry * ry + rz * rz;
  }
  __syncthreads();

  const int   esR = sEs[rowL], etR = sEt[rowL];
  const float sqR = sSq[rowL];

  // ---- GEMM1 -------------------------------------------------------------
  Acc acc1[16];
#pragma unroll
  for (int nb = 0; nb < 16; ++nb) {
    float bv = bm1[nb * 16 + laneLo];
#pragma unroll
    for (int r = 0; r < 8; ++r) acc1[nb].e[r] = bv;
  }
  for (int kb = 0; kb < 13; ++kb) {
    // stage W_m1 k-slab (32x256) into LDS in B-fragment order
    for (int c = tid; c < 256; c += 128) {
      int nb = c >> 4, c15 = c & 15;
      const __bf16* wrow = wm1 + (size_t)(kb * 32) * 256 + c;
#pragma unroll
      for (int r = 0; r < 32; ++r)
        sB[(((nb << 5) + c15 + (r & 16)) << 4) + (r & 15)] = wrow[(size_t)r * 256];
    }
    __syncthreads();
    // gather A fragment directly from feature tables (L2-hot)
    AFrag af;
#pragma unroll
    for (int h = 0; h < 16; ++h) {
      int f = kb * 32 + akmap(lane, h);
      float v;
      if      (f < 128) v = h_src[(size_t)esR * CCH + f];
      else if (f < 256) v = h_tgt[(size_t)etR * CCH + (f - 128)];
      else if (f == 256) v = sqR;
      else if (f < 385) v = t_emb[(size_t)etR * CCH + (f - 257)];
      else              v = 0.f;
      af.e[h] = (__bf16)v;
    }
#pragma unroll
    for (int nb = 0; nb < 16; ++nb) {
      v16bf bf = *(const v16bf*)(sB + (((nb << 5) + lane) << 4));
      acc1[nb].v = wmma_bf16(af.v, bf, acc1[nb].v);
    }
    __syncthreads();
  }
  __bf16* sHid = (__bf16*)sR1raw;           // 64 x 256 bf16
#pragma unroll
  for (int nb = 0; nb < 16; ++nb) {
    int N = nb * 16 + laneLo;
#pragma unroll
    for (int r = 0; r < 8; ++r) {
      int rl = wave * 16 + mBase + r;
      sHid[rl * 256 + N] = (__bf16)siluf(acc1[nb].e[r]);
    }
  }
  __syncthreads();

  // ---- GEMM2 -------------------------------------------------------------
  Acc acc2[8];
#pragma unroll
  for (int nb = 0; nb < 8; ++nb) {
    float bv = bm2[nb * 16 + laneLo];
#pragma unroll
    for (int r = 0; r < 8; ++r) acc2[nb].e[r] = bv;
  }
  for (int kb = 0; kb < 8; ++kb) {
    { // stage W_m2 k-slab (32x128)
      int nb = tid >> 4, c15 = tid & 15;
      const __bf16* wrow = wm2 + (size_t)(kb * 32) * 128 + tid;
#pragma unroll
      for (int r = 0; r < 32; ++r)
        sB[(((nb << 5) + c15 + (r & 16)) << 4) + (r & 15)] = wrow[(size_t)r * 128];
    }
    __syncthreads();
    AFrag af;
#pragma unroll
    for (int h = 0; h < 16; ++h)
      af.e[h] = sHid[rowL * 256 + kb * 32 + akmap(lane, h)];
#pragma unroll
    for (int nb = 0; nb < 8; ++nb) {
      v16bf bf = *(const v16bf*)(sB + (((nb << 5) + lane) << 4));
      acc2[nb].v = wmma_bf16(af.v, bf, acc2[nb].v);
    }
    __syncthreads();
  }
  // epilogue: silu -> msg ; scatter into agg ; keep bf16 msg tile in region0
  __bf16* sMsg = sB;                        // 64 x 128 bf16 (= 8192 entries)
#pragma unroll
  for (int nb = 0; nb < 8; ++nb) {
    int N = nb * 16 + laneLo;
#pragma unroll
    for (int r = 0; r < 8; ++r) {
      int rl = wave * 16 + mBase + r;
      float v = siluf(acc2[nb].e[r]);
      sMsg[rl * 128 + N] = (__bf16)v;
      atomicAdd(&agg[(size_t)sEt[rl] * CCH + N], v);
    }
  }
  __syncthreads();

  // ---- GEMM3 (gate): msg @ W_c1, B fragments straight from L2 ------------
  Acc acc3[4];
#pragma unroll
  for (int nb = 0; nb < 4; ++nb) {
    float bv = bc1[nb * 16 + laneLo];
#pragma unroll
    for (int r = 0; r < 8; ++r) acc3[nb].e[r] = bv;
  }
#pragma unroll
  for (int kb = 0; kb < 4; ++kb) {
    AFrag af;
#pragma unroll
    for (int h = 0; h < 16; ++h)
      af.e[h] = sMsg[rowL * 128 + kb * 32 + akmap(lane, h)];
#pragma unroll
    for (int nb = 0; nb < 4; ++nb) {
      AFrag bfr;
#pragma unroll
      for (int h = 0; h < 16; ++h)
        bfr.e[h] = wc1[(size_t)(kb * 32 + bkmap(lane, h)) * 64 + nb * 16 + laneLo];
      acc3[nb].v = wmma_bf16(af.v, bfr.v, acc3[nb].v);
    }
  }
  float* sH2 = (float*)sR1raw;              // 64 x 64 f32 (hid no longer needed)
#pragma unroll
  for (int nb = 0; nb < 4; ++nb) {
    int N = nb * 16 + laneLo;
#pragma unroll
    for (int r = 0; r < 8; ++r) {
      int rl = wave * 16 + mBase + r;
      sH2[rl * 64 + N] = siluf(acc3[nb].e[r]);
    }
  }
  __syncthreads();

  // ---- final dot with W_c2, tanh, vel scatter ----------------------------
  if (tid < 64) {
    float s = bc2[0];
#pragma unroll 8
    for (int k = 0; k < 64; ++k) s += sH2[tid * 64 + k] * wc2[k];
    float w = tanhf(s);
    int t = sEt[tid];
    atomicAdd(&vel[(size_t)t * 3 + 0], w * sRel[tid * 3 + 0]);
    atomicAdd(&vel[(size_t)t * 3 + 1], w * sRel[tid * 3 + 1]);
    atomicAdd(&vel[(size_t)t * 3 + 2], w * sRel[tid * 3 + 2]);
  }
}

// ---------------------------------------------------------------------------
// Node kernel: 64 target rows per block, 4 waves.
//   GEMM_u1: [64 x 256] ([h_tgt|agg]) @ [256 x 128] -> silu
//   GEMM_u2: [64 x 128] @ [128 x 128] ; y = h_tgt + upd ; LayerNorm -> out
// ---------------------------------------------------------------------------
__global__ __launch_bounds__(128) void egnn_node_kernel(
    const float* __restrict__ h_tgt, const float* __restrict__ agg,
    const __bf16* __restrict__ wu1, const float* __restrict__ bu1,
    const __bf16* __restrict__ wu2, const float* __restrict__ bu2,
    const float* __restrict__ gamma, const float* __restrict__ beta,
    float* __restrict__ out)
{
  __shared__ __align__(32) __bf16 sB[4096];             // 8 KB slab
  __shared__ __align__(32) unsigned char sR1raw[32768]; // hid bf16 / y f32

  const int tid    = threadIdx.x;
  const int lane   = tid & 31;
  const int wave   = tid >> 5;
  const int laneLo = lane & 15;
  const int mBase  = ((lane >> 4) << 3);
  const int rowL   = wave * 16 + laneLo;
  const int row0   = blockIdx.x * 64;
  const int growA  = row0 + rowL;
  const bool okA   = growA < N_TGT;

  // ---- GEMM_u1 -----------------------------------------------------------
  Acc a1[8];
#pragma unroll
  for (int nb = 0; nb < 8; ++nb) {
    float bv = bu1[nb * 16 + laneLo];
#pragma unroll
    for (int r = 0; r < 8; ++r) a1[nb].e[r] = bv;
  }
  for (int kb = 0; kb < 8; ++kb) {
    { int nb = tid >> 4, c15 = tid & 15;
      const __bf16* wrow = wu1 + (size_t)(kb * 32) * 128 + tid;
#pragma unroll
      for (int r = 0; r < 32; ++r)
        sB[(((nb << 5) + c15 + (r & 16)) << 4) + (r & 15)] = wrow[(size_t)r * 128];
    }
    __syncthreads();
    AFrag af;
#pragma unroll
    for (int h = 0; h < 16; ++h) {
      int f = kb * 32 + akmap(lane, h);
      float v = 0.f;
      if (okA)
        v = (f < 128) ? h_tgt[(size_t)growA * CCH + f]
                      : agg[(size_t)growA * CCH + (f - 128)];
      af.e[h] = (__bf16)v;
    }
#pragma unroll
    for (int nb = 0; nb < 8; ++nb) {
      v16bf bf = *(const v16bf*)(sB + (((nb << 5) + lane) << 4));
      a1[nb].v = wmma_bf16(af.v, bf, a1[nb].v);
    }
    __syncthreads();
  }
  __bf16* sHidN = (__bf16*)sR1raw;          // 64 x 128 bf16
#pragma unroll
  for (int nb = 0; nb < 8; ++nb) {
    int N = nb * 16 + laneLo;
#pragma unroll
    for (int r = 0; r < 8; ++r) {
      int rl = wave * 16 + mBase + r;
      sHidN[rl * 128 + N] = (__bf16)siluf(a1[nb].e[r]);
    }
  }
  __syncthreads();

  // ---- GEMM_u2 -----------------------------------------------------------
  Acc a2[8];
#pragma unroll
  for (int nb = 0; nb < 8; ++nb) {
    float bv = bu2[nb * 16 + laneLo];
#pragma unroll
    for (int r = 0; r < 8; ++r) a2[nb].e[r] = bv;
  }
  for (int kb = 0; kb < 4; ++kb) {
    { int nb = tid >> 4, c15 = tid & 15;
      const __bf16* wrow = wu2 + (size_t)(kb * 32) * 128 + tid;
#pragma unroll
      for (int r = 0; r < 32; ++r)
        sB[(((nb << 5) + c15 + (r & 16)) << 4) + (r & 15)] = wrow[(size_t)r * 128];
    }
    __syncthreads();
    AFrag af;
#pragma unroll
    for (int h = 0; h < 16; ++h)
      af.e[h] = sHidN[rowL * 128 + kb * 32 + akmap(lane, h)];
#pragma unroll
    for (int nb = 0; nb < 8; ++nb) {
      v16bf bf = *(const v16bf*)(sB + (((nb << 5) + lane) << 4));
      a2[nb].v = wmma_bf16(af.v, bf, a2[nb].v);
    }
    __syncthreads();
  }

  // ---- residual + LayerNorm ----------------------------------------------
  float* sOut = (float*)sR1raw;             // 64 x 128 f32 (overwrites sHidN)
#pragma unroll
  for (int nb = 0; nb < 8; ++nb) {
    int N = nb * 16 + laneLo;
#pragma unroll
    for (int r = 0; r < 8; ++r) {
      int rl = wave * 16 + mBase + r;
      int gr = row0 + rl;
      float h = (gr < N_TGT) ? h_tgt[(size_t)gr * CCH + N] : 0.f;
      sOut[rl * 128 + N] = a2[nb].e[r] + h;
    }
  }
  __syncthreads();
  if (tid < 64) {
    int gr = row0 + tid;
    if (gr < N_TGT) {
      const float* y = sOut + tid * 128;
      float mu = 0.f;
#pragma unroll 8
      for (int k = 0; k < 128; ++k) mu += y[k];
      mu *= (1.f / 128.f);
      float var = 0.f;
#pragma unroll 8
      for (int k = 0; k < 128; ++k) { float d = y[k] - mu; var += d * d; }
      var *= (1.f / 128.f);
      float inv = rsqrtf(var + 1e-5f);
#pragma unroll 8
      for (int k = 0; k < 128; ++k)
        out[(size_t)gr * CCH + k] = (y[k] - mu) * inv * gamma[k] + beta[k];
    }
  }
}

// ---------------------------------------------------------------------------
// Utility kernels
// ---------------------------------------------------------------------------
__global__ void zero_f32_kernel(float* p, long long n) {
  long long i = (long long)blockIdx.x * blockDim.x + threadIdx.x;
  if (i < n) p[i] = 0.f;
}
__global__ void cvt_pad_bf16_kernel(const float* __restrict__ s,
                                    __bf16* __restrict__ d,
                                    int rows, int cols, int srcRows) {
  int i = blockIdx.x * blockDim.x + threadIdx.x;
  if (i >= rows * cols) return;
  int r = i / cols, c = i - r * cols;
  d[i] = (__bf16)(r < srcRows ? s[r * cols + c] : 0.f);
}
__global__ void copy_f32_kernel(const float* __restrict__ s,
                                float* __restrict__ d, int n) {
  int i = blockIdx.x * blockDim.x + threadIdx.x;
  if (i < n) d[i] = s[i];
}

// ---------------------------------------------------------------------------
extern "C" void kernel_launch(void* const* d_in, const int* in_sizes, int n_in,
                              void* d_out, int out_size, void* d_ws, size_t ws_size,
                              hipStream_t stream) {
  (void)in_sizes; (void)n_in; (void)out_size; (void)ws_size;
  const float* h_src   = (const float*)d_in[0];
  const float* h_tgt   = (const float*)d_in[1];
  const float* pos_src = (const float*)d_in[2];
  const float* pos_tgt = (const float*)d_in[3];
  const float* t_emb   = (const float*)d_in[4];
  const float* W_m1 = (const float*)d_in[5];  const float* b_m1 = (const float*)d_in[6];
  const float* W_m2 = (const float*)d_in[7];  const float* b_m2 = (const float*)d_in[8];
  const float* W_c1 = (const float*)d_in[9];  const float* b_c1 = (const float*)d_in[10];
  const float* W_c2 = (const float*)d_in[11]; const float* b_c2 = (const float*)d_in[12];
  const float* W_u1 = (const float*)d_in[13]; const float* b_u1 = (const float*)d_in[14];
  const float* W_u2 = (const float*)d_in[15]; const float* b_u2 = (const float*)d_in[16];
  const float* gamma = (const float*)d_in[17]; const float* beta = (const float*)d_in[18];
  const int* edge_src = (const int*)d_in[19];
  const int* edge_tgt = (const int*)d_in[20];

  // workspace carve-out (agg and vel contiguous so one zero pass covers both)
  char* ws = (char*)d_ws;
  size_t o = 0;
  float* agg = (float*)(ws + o); o += (size_t)N_TGT * CCH * sizeof(float);
  float* vel = (float*)(ws + o); o += (size_t)N_TGT * 3 * sizeof(float);
  o = (o + 255) & ~(size_t)255;
  __bf16* wm1p = (__bf16*)(ws + o); o += (size_t)K1PAD * 256 * sizeof(__bf16);
  __bf16* wm2p = (__bf16*)(ws + o); o += (size_t)256 * 128 * sizeof(__bf16);
  __bf16* wc1p = (__bf16*)(ws + o); o += (size_t)128 * 64 * sizeof(__bf16);
  __bf16* wu1p = (__bf16*)(ws + o); o += (size_t)256 * 128 * sizeof(__bf16);
  __bf16* wu2p = (__bf16*)(ws + o); o += (size_t)128 * 128 * sizeof(__bf16);

  long long nz = (long long)N_TGT * (CCH + 3);
  zero_f32_kernel<<<(int)((nz + 255) / 256), 256, 0, stream>>>(agg, nz);

  cvt_pad_bf16_kernel<<<(K1PAD * 256 + 255) / 256, 256, 0, stream>>>(W_m1, wm1p, K1PAD, 256, 385);
  cvt_pad_bf16_kernel<<<(256 * 128 + 255) / 256, 256, 0, stream>>>(W_m2, wm2p, 256, 128, 256);
  cvt_pad_bf16_kernel<<<(128 * 64  + 255) / 256, 256, 0, stream>>>(W_c1, wc1p, 128, 64, 128);
  cvt_pad_bf16_kernel<<<(256 * 128 + 255) / 256, 256, 0, stream>>>(W_u1, wu1p, 256, 128, 256);
  cvt_pad_bf16_kernel<<<(128 * 128 + 255) / 256, 256, 0, stream>>>(W_u2, wu2p, 128, 128, 128);

  egnn_edge_kernel<<<N_EDGE / 64, 128, 0, stream>>>(
      h_src, h_tgt, pos_src, pos_tgt, t_emb,
      wm1p, b_m1, wm2p, b_m2, wc1p, b_c1, W_c2, b_c2,
      edge_src, edge_tgt, agg, vel);

  egnn_node_kernel<<<(N_TGT + 63) / 64, 128, 0, stream>>>(
      h_tgt, agg, wu1p, b_u1, wu2p, b_u2, gamma, beta, (float*)d_out);

  copy_f32_kernel<<<(N_TGT * 3 + 255) / 256, 256, 0, stream>>>(
      vel, (float*)d_out + (size_t)N_TGT * CCH, N_TGT * 3);
}
